// EvoformerBlock_36275293782043
// MI455X (gfx1250) — matrix operand order repaired and checked by reading, not compiled, not run on hardware
//
#include <hip/hip_runtime.h>
#include <math.h>
#include <stdint.h>

// ============================================================================
// Evoformer block for MI455X (gfx1250, wave32, WMMA).
//
// Precision strategy: fp32 storage everywhere; all contractions run on
// v_wmma_f32_16x16x32_f16 (f16 A/B fragments, fp32 accumulate). With ~125
// GFLOP of GEMM-shaped work vs ~100us of HBM traffic at 23.3 TB/s, the block
// is compute-limited without matrix cores, so every matmul-shaped op
// (projections, transitions, OPM, triangle einsums, QK^T / P*V) is WMMA.
//
// Data movement: A/K/V tiles are staged global->LDS with CDNA5 async copies
// (GLOBAL_LOAD_ASYNC_TO_LDS_B128, ASYNCcnt + s_wait_asynccnt), converting
// f32->f16 at fragment-build time.
//
// Input flattening assumption (jax tree_flatten, dict keys sorted at every
// level) -- documented, cannot be verified in compile-only mode:
//   0:m 1:pair_mask 2..85: params (opm,pt,sa,st,tae,tas,tmi,tmo; leaves
//   alphabetical within each) 86:seq_mask 87:z
// Output: m (81920 f32) then z (13107200 f32), concatenated.
// Workspace use: ~335 MB (6 z-sized pool buffers + small temps).
// ============================================================================

typedef _Float16 v16h __attribute__((ext_vector_type(16)));
typedef float    v8f  __attribute__((ext_vector_type(8)));

union AF { v16h v; _Float16 h[16]; };
union CF { v8f  v; float     f[8]; };

// Low 32 bits of a generic shared-memory address == LDS byte offset
// (CDNA5 ISA 10.2: LDS aperture).
static __device__ inline unsigned lds_off(const void* p) {
  return (unsigned)(uintptr_t)p;
}

// One 16-byte async global->LDS copy (tracked by ASYNCcnt).
static __device__ inline void async_copy_b128(unsigned lds, const float* gptr) {
  asm volatile("global_load_async_to_lds_b128 %0, %1, off"
               :: "v"(lds), "v"(gptr) : "memory");
}
static __device__ inline void async_wait0() {
  asm volatile("s_wait_asynccnt 0" ::: "memory");
}

// --------------------------------------------------------------------------
// LayerNorm over last dim. One wave per row, block = 8 waves.
// --------------------------------------------------------------------------
__global__ void __launch_bounds__(256) ln_kernel(
    const float* __restrict__ x, const float* __restrict__ g,
    const float* __restrict__ b, float* __restrict__ y, long long R, int C) {
  int wave = threadIdx.x >> 5, lane = threadIdx.x & 31;
  long long row = (long long)blockIdx.x * 8 + wave;
  if (row >= R) return;
  const float* xr = x + row * C;
  float s = 0.f, s2 = 0.f;
  for (int c = lane; c < C; c += 32) { float v = xr[c]; s += v; s2 += v * v; }
  for (int o = 1; o < 32; o <<= 1) {
    s  += __shfl_xor(s,  o, 32);
    s2 += __shfl_xor(s2, o, 32);
  }
  float mu  = s / C;
  float inv = rsqrtf(s2 / C - mu * mu + 1e-5f);
  float* yr = y + row * C;
  for (int c = lane; c < C; c += 32) yr[c] = (xr[c] - mu) * inv * g[c] + b[c];
}

// --------------------------------------------------------------------------
// Generic strided WMMA GEMM: C = epilogue(alpha * A@B).
// 4 waves/block; block tile 16(M) x 64(N); A tile staged in LDS (async b128
// when A is row-major contiguous, i.e. sAk==1 -- true for every call here),
// B fragments loaded directly from global with strides.
// Epilogue: +bias[col], act (0 none / 1 relu / 2 sigmoid), *gate[elem],
// *rowmask[row], +res[elem].
// Fragment layouts used here (contiguous k order for A and B):
//   A(16x32 f16): lane m=l&15, kh=l>>4; VGPR v -> K=kh*16 + 2v
//   B(32x16 f16): lane n=l&15, kh=l>>4; VGPR v -> K=kh*16 + 2v
//   C(16x16 f32): lane n=l&15; VGPR v -> row = v + 8*(l>>4)
// --------------------------------------------------------------------------
__global__ void __launch_bounds__(128) wmma_gemm_kernel(
    const float* __restrict__ A, const float* __restrict__ B,
    float* __restrict__ C,
    const float* __restrict__ bias, const float* __restrict__ gate,
    const float* __restrict__ rowmask, const float* __restrict__ res,
    int M, int N, int K, float alpha, int act,
    long long sAm, long long sAk, long long sBk, long long sBn, long long sCm) {
  __shared__ float Ash[16][32];
  const int wave = threadIdx.x >> 5;
  const int lane = threadIdx.x & 31;
  const int row0 = blockIdx.x * 16;
  const int n0   = blockIdx.y * 64 + wave * 16;
  const int nl   = lane & 15;
  const int kh   = lane >> 4;
  const int col  = n0 + nl;
  const long long bcol = (long long)((col < N) ? col : (N - 1));
  const bool fastA = (sAk == 1);  // wave-uniform

  CF acc;
#pragma unroll
  for (int v = 0; v < 8; ++v) acc.f[v] = 0.f;

  for (int k0 = 0; k0 < K; k0 += 32) {
    if (fastA) {
      // 16 rows x 128B: one async b128 chunk per thread.
      int r  = threadIdx.x >> 3;
      int c4 = (threadIdx.x & 7) << 2;
      async_copy_b128(lds_off(&Ash[r][c4]),
                      A + (long long)(row0 + r) * sAm + (k0 + c4));
      async_wait0();
    } else {
      for (int e = threadIdx.x; e < 512; e += 128) {
        int m = e >> 5, kk = e & 31;
        int gm = row0 + m;
        Ash[m][kk] = (gm < M)
            ? A[(long long)gm * sAm + (long long)(k0 + kk) * sAk] : 0.f;
      }
    }
    __syncthreads();
    AF af, bf;
#pragma unroll
    for (int v = 0; v < 8; ++v) {
      int kk = (kh << 4) + (v << 1);
      af.h[2 * v]     = (_Float16)Ash[nl][kk];
      af.h[2 * v + 1] = (_Float16)Ash[nl][kk + 1];
    }
#pragma unroll
    for (int v = 0; v < 8; ++v) {
      int kk = (kh << 4) + (v << 1);
      bf.h[2 * v]     = (_Float16)B[(long long)(k0 + kk) * sBk + bcol * sBn];
      bf.h[2 * v + 1] = (_Float16)B[(long long)(k0 + kk + 1) * sBk + bcol * sBn];
    }
    acc.v = __builtin_amdgcn_wmma_f32_16x16x32_f16(
        false, af.v, false, bf.v, (short)0, acc.v, false, false);
    __syncthreads();
  }

  if (col < N) {
#pragma unroll
    for (int v = 0; v < 8; ++v) {
      int row = row0 + v + (kh << 3);
      if (row < M) {
        long long ci = (long long)row * sCm + col;
        float val = alpha * acc.f[v] + (bias ? bias[col] : 0.f);
        if (act == 1)      val = fmaxf(val, 0.f);
        else if (act == 2) val = 1.f / (1.f + expf(-val));
        if (gate)    val *= gate[ci];
        if (rowmask) val *= rowmask[row];
        if (res)     val += res[ci];
        C[ci] = val;
      }
    }
  }
}

// --------------------------------------------------------------------------
// Fused pair-biased attention (flash-attention style, online softmax).
// Rows of Q/K/V/G/O are [i*Nq + q, H*32]. Bias[(q*Nk+k)*H + h].
// Mask[i*Nk + k] -> + 1e9*(mask-1).  G is pre-sigmoided gate.
// grid = (ceil(Nq/64), H, I); block = 128 (4 waves, 16 q-rows each).
// K/V tiles staged via async b128 copies; P re-laid-out through LDS to feed
// the P@V WMMA A-fragment. Row stats: S C-layout puts row m = v+8*(lane>>4)
// across the 16 lanes of one half-wave -> xor-shuffles over bits 0..3.
// --------------------------------------------------------------------------
__global__ void __launch_bounds__(128) fused_attn_kernel(
    const float* __restrict__ Q, const float* __restrict__ K,
    const float* __restrict__ V, const float* __restrict__ G,
    const float* __restrict__ Bias, const float* __restrict__ Mask,
    float* __restrict__ O, int I, int Nq, int Nk, int H) {
  const int ld   = H * 32;
  const int h    = blockIdx.y;
  const int iI   = blockIdx.z;
  const int wave = threadIdx.x >> 5;
  const int lane = threadIdx.x & 31;
  const int q0   = blockIdx.x * 64 + wave * 16;

  __shared__ float    Ksh[32][32];
  __shared__ float    Vsh[32][32];
  __shared__ _Float16 Psh[4][16][32];

  const int n  = lane & 15;
  const int ch = lane >> 4;

  AF qf;
  {
    const float* qp = Q + ((long long)iI * Nq + q0 + n) * ld + h * 32;
#pragma unroll
    for (int v = 0; v < 8; ++v) {
      int c = (ch << 4) + (v << 1);
      qf.h[2 * v]     = (_Float16)qp[c];
      qf.h[2 * v + 1] = (_Float16)qp[c + 1];
    }
  }

  float rm[8], rl[8];
  CF acc0, acc1;
#pragma unroll
  for (int v = 0; v < 8; ++v) { rm[v] = -1e30f; rl[v] = 0.f; acc0.f[v] = 0.f; acc1.f[v] = 0.f; }

  for (int k0 = 0; k0 < Nk; k0 += 32) {
    {
      // 32 rows x 128B for K and V: two async b128 chunks per thread each.
      int r  = threadIdx.x >> 3;        // 0..15
      int c4 = (threadIdx.x & 7) << 2;  // 0,4,...,28
      long long base = ((long long)iI * Nk + k0) * ld + h * 32 + c4;
      async_copy_b128(lds_off(&Ksh[r][c4]),      K + base + (long long)r * ld);
      async_copy_b128(lds_off(&Ksh[r + 16][c4]), K + base + (long long)(r + 16) * ld);
      async_copy_b128(lds_off(&Vsh[r][c4]),      V + base + (long long)r * ld);
      async_copy_b128(lds_off(&Vsh[r + 16][c4]), V + base + (long long)(r + 16) * ld);
      async_wait0();
    }
    __syncthreads();

    // scores: S = Q (16x32c) @ K^T (32c x 32k) as two 16x16 WMMAs
    AF b0, b1;
#pragma unroll
    for (int v = 0; v < 8; ++v) {
      int c = (ch << 4) + (v << 1);
      b0.h[2 * v]     = (_Float16)Ksh[n][c];
      b0.h[2 * v + 1] = (_Float16)Ksh[n][c + 1];
      b1.h[2 * v]     = (_Float16)Ksh[16 + n][c];
      b1.h[2 * v + 1] = (_Float16)Ksh[16 + n][c + 1];
    }
    CF zc;
#pragma unroll
    for (int v = 0; v < 8; ++v) zc.f[v] = 0.f;
    CF s0, s1;
    s0.v = __builtin_amdgcn_wmma_f32_16x16x32_f16(false, qf.v, false, b0.v, (short)0, zc.v, false, false);
    s1.v = __builtin_amdgcn_wmma_f32_16x16x32_f16(false, qf.v, false, b1.v, (short)0, zc.v, false, false);

    float mk0 = 1e9f * (Mask[(long long)iI * Nk + k0 + n] - 1.f);
    float mk1 = 1e9f * (Mask[(long long)iI * Nk + k0 + 16 + n] - 1.f);
#pragma unroll
    for (int v = 0; v < 8; ++v) {
      int m = v + (ch << 3);
      int q = q0 + m;
      float e0 = s0.f[v] + Bias[((long long)q * Nk + k0 + n) * H + h] + mk0;
      float e1 = s1.f[v] + Bias[((long long)q * Nk + k0 + 16 + n) * H + h] + mk1;
      float mx = fmaxf(e0, e1);
      mx = fmaxf(mx, __shfl_xor(mx, 1, 32));
      mx = fmaxf(mx, __shfl_xor(mx, 2, 32));
      mx = fmaxf(mx, __shfl_xor(mx, 4, 32));
      mx = fmaxf(mx, __shfl_xor(mx, 8, 32));
      float nm   = fmaxf(rm[v], mx);
      float corr = expf(rm[v] - nm);
      rm[v] = nm;
      float p0 = expf(e0 - nm), p1 = expf(e1 - nm);
      float ps = p0 + p1;
      ps += __shfl_xor(ps, 1, 32);
      ps += __shfl_xor(ps, 2, 32);
      ps += __shfl_xor(ps, 4, 32);
      ps += __shfl_xor(ps, 8, 32);
      rl[v] = rl[v] * corr + ps;
      acc0.f[v] *= corr;
      acc1.f[v] *= corr;
      Psh[wave][m][n]      = (_Float16)p0;
      Psh[wave][m][16 + n] = (_Float16)p1;
    }

    // O += P (16x32k) @ V (32k x 32c) as two 16x16 WMMAs
    AF pf;
#pragma unroll
    for (int v = 0; v < 8; ++v) {
      int kk = (ch << 4) + (v << 1);
      pf.h[2 * v]     = Psh[wave][n][kk];
      pf.h[2 * v + 1] = Psh[wave][n][kk + 1];
    }
    AF bv0, bv1;
#pragma unroll
    for (int v = 0; v < 8; ++v) {
      int kk = (ch << 4) + (v << 1);
      bv0.h[2 * v]     = (_Float16)Vsh[kk][n];
      bv0.h[2 * v + 1] = (_Float16)Vsh[kk + 1][n];
      bv1.h[2 * v]     = (_Float16)Vsh[kk][16 + n];
      bv1.h[2 * v + 1] = (_Float16)Vsh[kk + 1][16 + n];
    }
    acc0.v = __builtin_amdgcn_wmma_f32_16x16x32_f16(false, pf.v, false, bv0.v, (short)0, acc0.v, false, false);
    acc1.v = __builtin_amdgcn_wmma_f32_16x16x32_f16(false, pf.v, false, bv1.v, (short)0, acc1.v, false, false);
    __syncthreads();
  }

#pragma unroll
  for (int v = 0; v < 8; ++v) {
    int q = q0 + v + (ch << 3);
    long long base = ((long long)iI * Nq + q) * ld + h * 32;
    float inv = 1.f / rl[v];
    O[base + n]      = acc0.f[v] * inv * G[base + n];
    O[base + 16 + n] = acc1.f[v] * inv * G[base + 16 + n];
  }
}

// --------------------------------------------------------------------------
// Triangle-multiplication einsum, batched over the 128 channels:
//   outgoing: X[i,j,c] = sum_k A[i,k,c] * B[j,k,c]
//   incoming: X[i,j,c] = sum_k A[k,i,c] * B[k,j,c]
// grid = (20 i-tiles, 20 j-tiles, 128 channels); 1 wave per block.
// Channel-strided gathers -> direct per-lane loads (no useful vector shape).
// --------------------------------------------------------------------------
__global__ void __launch_bounds__(32) tri_einsum_kernel(
    const float* __restrict__ A, const float* __restrict__ Bm,
    float* __restrict__ X, int outgoing) {
  const int lane = threadIdx.x;
  const int i0 = blockIdx.x * 16;
  const int j0 = blockIdx.y * 16;
  const int c  = blockIdx.z;
  const int n  = lane & 15;
  const int kh = lane >> 4;
  CF acc;
#pragma unroll
  for (int v = 0; v < 8; ++v) acc.f[v] = 0.f;

  for (int k0 = 0; k0 < 320; k0 += 32) {
    AF af, bf;
#pragma unroll
    for (int v = 0; v < 8; ++v) {
      int ka = (kh << 4) + (v << 1);
      long long ia, sa;
      if (outgoing) { ia = (long long)(i0 + n) * 320 + (k0 + ka); sa = 1; }
      else          { ia = (long long)(k0 + ka) * 320 + (i0 + n); sa = 320; }
      af.h[2 * v]     = (_Float16)A[ia * 128 + c];
      af.h[2 * v + 1] = (_Float16)A[(ia + sa) * 128 + c];
      int kb = (kh << 4) + (v << 1);
      long long ib, sb;
      if (outgoing) { ib = (long long)(j0 + n) * 320 + (k0 + kb); sb = 1; }
      else          { ib = (long long)(k0 + kb) * 320 + (j0 + n); sb = 320; }
      bf.h[2 * v]     = (_Float16)Bm[ib * 128 + c];
      bf.h[2 * v + 1] = (_Float16)Bm[(ib + sb) * 128 + c];
    }
    acc.v = __builtin_amdgcn_wmma_f32_16x16x32_f16(
        false, af.v, false, bf.v, (short)0, acc.v, false, false);
  }
#pragma unroll
  for (int v = 0; v < 8; ++v) {
    int row = i0 + v + (kh << 3);
    X[((long long)row * 320 + (j0 + n)) * 128 + c] = acc.f[v];
  }
}

// --------------------------------------------------------------------------
// Elementwise helpers
// --------------------------------------------------------------------------
__global__ void ew_mul3_kernel(float* out, const float* xx, const float* yy,
                               const float* __restrict__ rowmask,
                               long long nTot, int C) {
  long long i = (long long)blockIdx.x * 256 + threadIdx.x;
  if (i < nTot) out[i] = rowmask[i / C] * xx[i] * yy[i];
}

__global__ void opm_finish_kernel(float* __restrict__ zout,
                                  const float* __restrict__ zin,
                                  const float* __restrict__ o,
                                  const float* __restrict__ bo,
                                  const float* __restrict__ sm) {
  long long i = (long long)blockIdx.x * 256 + threadIdx.x;
  const long long total = 102400LL * 128;
  if (i >= total) return;
  int c = (int)(i & 127);
  long long pij = i >> 7;
  int jj = (int)(pij % 320);
  int ii = (int)(pij / 320);
  float norm = sm[ii] * sm[jj];
  zout[i] = zin[i] + (o[i] + bo[c]) / (1e-3f + norm);
}

// out[(j,i,c)] = in[(i,j,c)] + (add ? add[(j,i,c)] : 0)
__global__ void transpose_kernel(float* __restrict__ out,
                                 const float* __restrict__ in,
                                 const float* add, int N, int C) {
  long long i = (long long)blockIdx.x * 256 + threadIdx.x;
  long long total = (long long)N * N * C;
  if (i >= total) return;
  int c = (int)(i % C);
  long long p = i / C;
  int jj = (int)(p % N);
  int ii = (int)(p / N);
  long long oi = ((long long)jj * N + ii) * C + c;
  float v = in[i];
  out[oi] = add ? (v + add[oi]) : v;
}

// W'[d, c*128+o] = wo[c*32+d, o]   (OPM factoring)
__global__ void permute_wo_kernel(float* __restrict__ dst,
                                  const float* __restrict__ src) {
  int i = blockIdx.x * 256 + threadIdx.x;  // 32*32*128
  if (i >= 131072) return;
  int o = i & 127; int r = i >> 7; int c = r & 31; int d = r >> 5;
  dst[(d * 32 + c) * 128 + o] = src[(c * 32 + d) * 128 + o];
}

// T2[c, j*128+o] = T[j, c*128+o]
__global__ void permute_T_kernel(float* __restrict__ dst,
                                 const float* __restrict__ src) {
  int i = blockIdx.x * 256 + threadIdx.x;  // 320*32*128
  if (i >= 1310720) return;
  int o = i & 127; int r = i >> 7; int j = r % 320; int c = r / 320;
  dst[((long long)c * 320 + j) * 128 + o] = src[((long long)j * 32 + c) * 128 + o];
}

// ============================================================================
// Host orchestration
// ============================================================================
enum {
  IN_M = 0, IN_PAIRMASK = 1,
  OPM_BA = 2, OPM_BB, OPM_BO, OPM_LN_B, OPM_LN_G, OPM_WA, OPM_WB, OPM_WO,
  PT_B1 = 10, PT_B2, PT_LN_B, PT_LN_G, PT_W1, PT_W2,
  SA_BG = 16, SA_BO, SA_LNM_B, SA_LNM_G, SA_LNZ_B, SA_LNZ_G,
  SA_WG, SA_WK, SA_WO, SA_WQ, SA_WV, SA_WZ,
  ST_B1 = 28, ST_B2, ST_LN_B, ST_LN_G, ST_W1, ST_W2,
  TAE_BG = 34, TAE_BO, TAE_LN_B, TAE_LN_G, TAE_WB, TAE_WG, TAE_WK, TAE_WO, TAE_WQ, TAE_WV,
  TAS_BG = 44, TAS_BO, TAS_LN_B, TAS_LN_G, TAS_WB, TAS_WG, TAS_WK, TAS_WO, TAS_WQ, TAS_WV,
  TMI_BAG = 54, TMI_BAP, TMI_BBG, TMI_BBP, TMI_BG, TMI_BZ,
  TMI_LNIN_B, TMI_LNIN_G, TMI_LNOUT_B, TMI_LNOUT_G,
  TMI_WAG, TMI_WAP, TMI_WBG, TMI_WBP, TMI_WG, TMI_WZ,
  TMO_BAG = 70, TMO_BAP, TMO_BBG, TMO_BBP, TMO_BG, TMO_BZ,
  TMO_LNIN_B, TMO_LNIN_G, TMO_LNOUT_B, TMO_LNOUT_G,
  TMO_WAG, TMO_WAP, TMO_WBG, TMO_WBP, TMO_WG, TMO_WZ,
  IN_SEQMASK = 86, IN_Z = 87
};

extern "C" void kernel_launch(void* const* d_in, const int* in_sizes, int n_in,
                              void* d_out, int out_size, void* d_ws, size_t ws_size,
                              hipStream_t stream) {
  (void)in_sizes; (void)n_in; (void)out_size; (void)ws_size;
  auto P = [&](int i) { return (const float*)d_in[i]; };

  const long long NN = 102400;       // 320*320
  const long long ZSZ = NN * 128;    // 13,107,200 floats
  const float INVS = 0.17677669529663687f;  // 1/sqrt(32)

  float* W = (float*)d_ws;
  size_t off = 0;
  auto alloc = [&](size_t nf) { float* p = W + off; off += nf; return p; };
  float* ZP[6];
  for (int i = 0; i < 6; ++i) ZP[i] = alloc((size_t)ZSZ);
  float* MB    = alloc(81920);
  float* QB    = alloc(81920);
  float* KB    = alloc(81920);
  float* VB    = alloc(81920);
  float* GB    = alloc(81920);
  float* OB    = alloc(81920);
  float* MC0   = alloc(81920);
  float* MC1   = alloc(81920);
  float* H1B   = alloc(327680);
  float* ZBIAS = alloc(819200);   // [q,k,8]
  float* A32   = alloc(10240);
  float* B32   = alloc(10240);
  float* WP    = alloc(131072);   // [32,4096]
  float* TB    = alloc(1310720);  // [320,4096]
  float* T2B   = alloc(1310720);  // [32,40960]
  float* TRIB  = alloc(409600);   // [q,k,4]
  float* MASKT = alloc(102400);

  const float* m_in  = P(IN_M);
  const float* pmask = P(IN_PAIRMASK);
  const float* smask = P(IN_SEQMASK);
  const float* z_in  = P(IN_Z);

  auto ln = [&](const float* x, const float* g, const float* b, float* y,
                long long R, int C) {
    ln_kernel<<<dim3((unsigned)((R + 7) / 8)), 256, 0, stream>>>(x, g, b, y, R, C);
  };
  auto gemm = [&](const float* A, const float* Bw, float* Co,
                  const float* bias, const float* gate, const float* rowmask,
                  const float* res, int M, int N, int K, float alpha, int act) {
    dim3 g((M + 15) / 16, (N + 63) / 64, 1);
    wmma_gemm_kernel<<<g, 128, 0, stream>>>(A, Bw, Co, bias, gate, rowmask, res,
                                            M, N, K, alpha, act,
                                            (long long)K, 1LL, (long long)N, 1LL,
                                            (long long)N);
  };
  const long long ZBLK = (ZSZ + 255) / 256;

  // ---------------- 1) sequence attention (uses original m, z) -------------
  ln(m_in, P(SA_LNM_G), P(SA_LNM_B), MB, 320, 256);
  ln(z_in, P(SA_LNZ_G), P(SA_LNZ_B), ZP[0], NN, 128);
  gemm(MB, P(SA_WQ), QB, nullptr, nullptr, nullptr, nullptr, 320, 256, 256, INVS, 0);
  gemm(MB, P(SA_WK), KB, nullptr, nullptr, nullptr, nullptr, 320, 256, 256, 1.f, 0);
  gemm(MB, P(SA_WV), VB, nullptr, nullptr, nullptr, nullptr, 320, 256, 256, 1.f, 0);
  gemm(MB, P(SA_WG), GB, P(SA_BG), nullptr, nullptr, nullptr, 320, 256, 256, 1.f, 2);
  gemm(ZP[0], P(SA_WZ), ZBIAS, nullptr, nullptr, nullptr, nullptr, (int)NN, 8, 128, 1.f, 0);
  fused_attn_kernel<<<dim3(5, 8, 1), 128, 0, stream>>>(
      QB, KB, VB, GB, ZBIAS, smask, OB, 1, 320, 320, 8);
  gemm(OB, P(SA_WO), MC0, P(SA_BO), nullptr, nullptr, m_in, 320, 256, 256, 1.f, 0);

  // ---------------- 2) sequence transition ---------------------------------
  ln(MC0, P(ST_LN_G), P(ST_LN_B), MB, 320, 256);
  gemm(MB, P(ST_W1), H1B, P(ST_B1), nullptr, nullptr, nullptr, 320, 1024, 256, 1.f, 1);
  gemm(H1B, P(ST_W2), MC1, P(ST_B2), nullptr, smask, MC0, 320, 256, 1024, 1.f, 0);

  // ---------------- 3) outer product mean (factored) ------------------------
  ln(MC1, P(OPM_LN_G), P(OPM_LN_B), MB, 320, 256);
  gemm(MB, P(OPM_WA), A32, P(OPM_BA), nullptr, smask, nullptr, 320, 32, 256, 1.f, 0);
  gemm(MB, P(OPM_WB), B32, P(OPM_BB), nullptr, smask, nullptr, 320, 32, 256, 1.f, 0);
  permute_wo_kernel<<<512, 256, 0, stream>>>(WP, P(OPM_WO));
  gemm(B32, WP, TB, nullptr, nullptr, nullptr, nullptr, 320, 4096, 32, 1.f, 0);
  permute_T_kernel<<<5120, 256, 0, stream>>>(T2B, TB);
  gemm(A32, T2B, ZP[1], nullptr, nullptr, nullptr, nullptr, 320, 40960, 32, 1.f, 0);
  opm_finish_kernel<<<(unsigned)ZBLK, 256, 0, stream>>>(ZP[0], z_in, ZP[1],
                                                        P(OPM_BO), smask);
  // zcur = ZP[0]

  // ---------------- 4) triangle multiply, outgoing --------------------------
  ln(ZP[0], P(TMO_LNIN_G), P(TMO_LNIN_B), ZP[1], NN, 128);
  gemm(ZP[1], P(TMO_WAG), ZP[2], P(TMO_BAG), nullptr, nullptr, nullptr, (int)NN, 128, 128, 1.f, 2);
  gemm(ZP[1], P(TMO_WAP), ZP[3], P(TMO_BAP), nullptr, nullptr, nullptr, (int)NN, 128, 128, 1.f, 0);
  ew_mul3_kernel<<<(unsigned)ZBLK, 256, 0, stream>>>(ZP[2], ZP[2], ZP[3], pmask, ZSZ, 128);
  gemm(ZP[1], P(TMO_WBG), ZP[3], P(TMO_BBG), nullptr, nullptr, nullptr, (int)NN, 128, 128, 1.f, 2);
  gemm(ZP[1], P(TMO_WBP), ZP[4], P(TMO_BBP), nullptr, nullptr, nullptr, (int)NN, 128, 128, 1.f, 0);
  ew_mul3_kernel<<<(unsigned)ZBLK, 256, 0, stream>>>(ZP[3], ZP[3], ZP[4], pmask, ZSZ, 128);
  gemm(ZP[1], P(TMO_WG), ZP[4], P(TMO_BG), nullptr, nullptr, nullptr, (int)NN, 128, 128, 1.f, 2);
  tri_einsum_kernel<<<dim3(20, 20, 128), 32, 0, stream>>>(ZP[2], ZP[3], ZP[1], 1);
  ln(ZP[1], P(TMO_LNOUT_G), P(TMO_LNOUT_B), ZP[2], NN, 128);
  gemm(ZP[2], P(TMO_WZ), ZP[3], P(TMO_BZ), ZP[4], nullptr, ZP[0], (int)NN, 128, 128, 1.f, 0);
  // zcur = ZP[3]

  // ---------------- 5) triangle multiply, incoming ---------------------------
  ln(ZP[3], P(TMI_LNIN_G), P(TMI_LNIN_B), ZP[0], NN, 128);
  gemm(ZP[0], P(TMI_WAG), ZP[1], P(TMI_BAG), nullptr, nullptr, nullptr, (int)NN, 128, 128, 1.f, 2);
  gemm(ZP[0], P(TMI_WAP), ZP[2], P(TMI_BAP), nullptr, nullptr, nullptr, (int)NN, 128, 128, 1.f, 0);
  ew_mul3_kernel<<<(unsigned)ZBLK, 256, 0, stream>>>(ZP[1], ZP[1], ZP[2], pmask, ZSZ, 128);
  gemm(ZP[0], P(TMI_WBG), ZP[2], P(TMI_BBG), nullptr, nullptr, nullptr, (int)NN, 128, 128, 1.f, 2);
  gemm(ZP[0], P(TMI_WBP), ZP[4], P(TMI_BBP), nullptr, nullptr, nullptr, (int)NN, 128, 128, 1.f, 0);
  ew_mul3_kernel<<<(unsigned)ZBLK, 256, 0, stream>>>(ZP[2], ZP[2], ZP[4], pmask, ZSZ, 128);
  gemm(ZP[0], P(TMI_WG), ZP[4], P(TMI_BG), nullptr, nullptr, nullptr, (int)NN, 128, 128, 1.f, 2);
  tri_einsum_kernel<<<dim3(20, 20, 128), 32, 0, stream>>>(ZP[1], ZP[2], ZP[0], 0);
  ln(ZP[0], P(TMI_LNOUT_G), P(TMI_LNOUT_B), ZP[1], NN, 128);
  gemm(ZP[1], P(TMI_WZ), ZP[5], P(TMI_BZ), ZP[4], nullptr, ZP[3], (int)NN, 128, 128, 1.f, 0);
  // zcur = ZP[5]

  // ---------------- 6) triangle attention, starting node ---------------------
  ln(ZP[5], P(TAS_LN_G), P(TAS_LN_B), ZP[0], NN, 128);
  gemm(ZP[0], P(TAS_WQ), ZP[1], nullptr, nullptr, nullptr, nullptr, (int)NN, 128, 128, INVS, 0);
  gemm(ZP[0], P(TAS_WK), ZP[2], nullptr, nullptr, nullptr, nullptr, (int)NN, 128, 128, 1.f, 0);
  gemm(ZP[0], P(TAS_WV), ZP[3], nullptr, nullptr, nullptr, nullptr, (int)NN, 128, 128, 1.f, 0);
  gemm(ZP[0], P(TAS_WG), ZP[4], P(TAS_BG), nullptr, nullptr, nullptr, (int)NN, 128, 128, 1.f, 2);
  gemm(ZP[0], P(TAS_WB), TRIB, nullptr, nullptr, nullptr, nullptr, (int)NN, 4, 128, 1.f, 0);
  fused_attn_kernel<<<dim3(5, 4, 320), 128, 0, stream>>>(
      ZP[1], ZP[2], ZP[3], ZP[4], TRIB, pmask, ZP[0], 320, 320, 320, 4);
  gemm(ZP[0], P(TAS_WO), ZP[1], P(TAS_BO), nullptr, nullptr, ZP[5], (int)NN, 128, 128, 1.f, 0);
  // zcur = ZP[1]

  // ---------------- 7) triangle attention, ending node (on z^T) --------------
  transpose_kernel<<<(unsigned)ZBLK, 256, 0, stream>>>(ZP[0], ZP[1], nullptr, 320, 128);
  transpose_kernel<<<(102400 + 255) / 256, 256, 0, stream>>>(MASKT, pmask, nullptr, 320, 1);
  ln(ZP[0], P(TAE_LN_G), P(TAE_LN_B), ZP[2], NN, 128);
  gemm(ZP[2], P(TAE_WQ), ZP[0], nullptr, nullptr, nullptr, nullptr, (int)NN, 128, 128, INVS, 0);
  gemm(ZP[2], P(TAE_WK), ZP[3], nullptr, nullptr, nullptr, nullptr, (int)NN, 128, 128, 1.f, 0);
  gemm(ZP[2], P(TAE_WV), ZP[4], nullptr, nullptr, nullptr, nullptr, (int)NN, 128, 128, 1.f, 0);
  gemm(ZP[2], P(TAE_WG), ZP[5], P(TAE_BG), nullptr, nullptr, nullptr, (int)NN, 128, 128, 1.f, 2);
  gemm(ZP[2], P(TAE_WB), TRIB, nullptr, nullptr, nullptr, nullptr, (int)NN, 4, 128, 1.f, 0);
  fused_attn_kernel<<<dim3(5, 4, 320), 128, 0, stream>>>(
      ZP[0], ZP[3], ZP[4], ZP[5], TRIB, MASKT, ZP[2], 320, 320, 320, 4);
  gemm(ZP[2], P(TAE_WO), ZP[0], P(TAE_BO), nullptr, nullptr, nullptr, (int)NN, 128, 128, 1.f, 0);
  // z = zcur + r^T : out[(j,i)] = r[(i,j)] + zcur[(j,i)]
  transpose_kernel<<<(unsigned)ZBLK, 256, 0, stream>>>(ZP[2], ZP[0], ZP[1], 320, 128);
  // zcur = ZP[2]

  // ---------------- 8) pair transition (chunked, 4 x 25600 rows) -------------
  ln(ZP[2], P(PT_LN_G), P(PT_LN_B), ZP[0], NN, 128);
  for (int chunk = 0; chunk < 4; ++chunk) {
    long long r0 = (long long)chunk * 25600;
    gemm(ZP[0] + r0 * 128, P(PT_W1), ZP[1], P(PT_B1), nullptr, nullptr, nullptr,
         25600, 512, 128, 1.f, 1);
    gemm(ZP[1], P(PT_W2), ZP[3] + r0 * 128, P(PT_B2), nullptr, pmask + r0,
         ZP[2] + r0 * 128, 25600, 128, 512, 1.f, 0);
  }
  // zcur = ZP[3]

  // ---------------- outputs: (m, z) concatenated -----------------------------
  hipMemcpyAsync(d_out, MC1, 81920 * sizeof(float),
                 hipMemcpyDeviceToDevice, stream);
  hipMemcpyAsync((float*)d_out + 81920, ZP[3], (size_t)ZSZ * sizeof(float),
                 hipMemcpyDeviceToDevice, stream);
}
